// MultiRelationGraphTransformer_79018808312384
// MI455X (gfx1250) — matrix operand order repaired and checked by reading, not compiled, not run on hardware
//
#include <hip/hip_runtime.h>
#include <hip/hip_bf16.h>
#include <cstdint>

typedef __attribute__((ext_vector_type(16))) _Float16 v16h;
typedef __attribute__((ext_vector_type(8)))  float    v8f;

#define B_  16
#define N_  1024
#define D_  256
#define F_  128
#define L_  2
#define R_  2
#define DF_ 384          // D + F
#define RD_ 512          // R * D

// ---------------------------------------------------------------------------
// CDNA5 async copy: global -> LDS, 16B per lane. VDST holds the LDS byte
// address (flat LDS pointers carry the wave-relative offset in addr[31:0],
// ISA 10.2 aperture truncation). Tracked by ASYNCcnt.
// ---------------------------------------------------------------------------
__device__ __forceinline__ void async_ld_b128(void* ldsp, const void* gp) {
  uint32_t loff = (uint32_t)(uintptr_t)ldsp;
  asm volatile("global_load_async_to_lds_b128 %0, %1, off"
               :: "v"(loff), "v"(gp)
               : "memory");
}
__device__ __forceinline__ void wait_async0() {
  asm volatile("s_wait_asynccnt 0x0" ::: "memory");
}

// ---------------------------------------------------------------------------
// WMMA fragment loaders (layouts per CDNA5 ISA 7.12.2, wave32)
// ---------------------------------------------------------------------------

// A-matrix 16x32 f16, row-major source, leading dim ld (halves).
__device__ __forceinline__ v16h load_a16(const _Float16* __restrict__ p, int ld, int lane) {
  const int m = lane & 15;
  const int g = lane >> 4;
  const _Float16* p0 = p + (size_t)m * ld + g * 8;
  v16h a;
#pragma unroll
  for (int i = 0; i < 8; ++i) { a[i] = p0[i]; a[i + 8] = p0[16 + i]; }
  return a;
}

// A-matrix 16x32 from f32 LDS scores (convert to f16 on load).
__device__ __forceinline__ v16h load_a_lds32(const float* __restrict__ p, int ld, int lane) {
  const int m = lane & 15;
  const int g = lane >> 4;
  const float* p0 = p + (size_t)m * ld + g * 8;
  v16h a;
#pragma unroll
  for (int i = 0; i < 8; ++i) {
    a[i]     = (_Float16)p0[i];
    a[i + 8] = (_Float16)p0[16 + i];
  }
  return a;
}

// B-matrix 32x16 f16 from transposed ("BT", Nout x K row-major) source
// (works for global or LDS pointers). Two 16B chunks per lane (k 8h..8h+7, 16+8h..16+8h+7).
__device__ __forceinline__ v16h load_bt16(const _Float16* bt, int ld, int lane) {
  const int n  = lane & 15;
  const int kb = (lane >> 4) * 8;
  const _Float16* p = bt + (size_t)n * ld + kb;
  v16h b;
#pragma unroll
  for (int i = 0; i < 8; ++i) { b[i] = p[i]; b[i + 8] = p[16 + i]; }
  return b;
}

__device__ __forceinline__ v8f wmma_f16(v16h a, v16h b, v8f c) {
  return __builtin_amdgcn_wmma_f32_16x16x32_f16(false, a, false, b, (short)0, c, false, false);
}

// ---------------------------------------------------------------------------
// Small prep kernels
// ---------------------------------------------------------------------------

__global__ void x1_kernel(const float* __restrict__ x, const float* __restrict__ w,
                          const float* __restrict__ bias, float* __restrict__ x1) {
  const int b = blockIdx.x, f = threadIdx.x;
  float acc = bias[f];
  for (int j = 0; j < F_; ++j) acc += x[b * F_ + j] * w[j * F_ + f];
  x1[b * F_ + f] = acc;
}

__global__ void build_qc_kernel(const float* __restrict__ h, const float* __restrict__ x1,
                                _Float16* __restrict__ qc) {
  const size_t bn = blockIdx.x;
  const int t = threadIdx.x;
  const int b = (int)(bn / N_);
  const float v = (t < D_) ? h[bn * D_ + t] : x1[b * F_ + (t - D_)];
  qc[bn * DF_ + t] = (_Float16)v;
}

// wt[n,k] = (f16) w[k,n], w is (K, 256) row-major.
__global__ void tconv_kernel(const float* __restrict__ w, _Float16* __restrict__ wt, int K) {
  const size_t idx = (size_t)blockIdx.x * blockDim.x + threadIdx.x;
  const size_t total = (size_t)K * 256;
  if (idx >= total) return;
  const int n = (int)(idx / K);
  const int k = (int)(idx % K);
  wt[idx] = (_Float16)w[(size_t)k * 256 + n];
}

// ---------------------------------------------------------------------------
// WMMA GEMM:  C(M x 256) = A(M x K) @ WT^T + bias.
// WT stored (256 x K) row-major; the block's 64-column slab of WT is staged
// into LDS with global_load_async_to_lds_b128 and shared by all 8 waves.
// grid(M/128, 4), block(256): wave -> 16 rows x 64 cols (4 tiles, 1 A-load
// feeds 4 WMMA; all 4 B fragments loaded up-front per K-step so the ds_loads
// batch into one clause/wait). Dynamic LDS = 64*K*2 bytes.
// mode 0: f16 row-major out (ld 256)
// mode 1: f32 row-major out + bias + ReLU
// mode 2: f16 transposed out -> vT[((row/N)*D + col)*N + row%N] (b128 stores)
// ---------------------------------------------------------------------------
__global__ void wmma_gemm_kernel(const _Float16* __restrict__ A,
                                 const _Float16* __restrict__ WT,
                                 const float*    __restrict__ bias,
                                 void* __restrict__ outp,
                                 int K, int mode) {
  extern __shared__ _Float16 smW[];            // 64 x K tile of WT
  const int tid  = threadIdx.x;
  const int lane = tid & 31;
  const int w    = tid >> 5;
  const int row0 = blockIdx.x * 128 + w * 16;
  const int col0 = blockIdx.y * 64;

  // ---- async stage of the WT slab (rows col0..col0+63 are contiguous) ----
  {
    const char* gsrc = (const char*)(WT + (size_t)col0 * K);
    char* ldst = (char*)smW;
    const int n16 = (64 * K * 2) >> 4;         // 16B chunks
    for (int i = tid; i < n16; i += 256)
      async_ld_b128(ldst + (size_t)i * 16, gsrc + (size_t)i * 16);
    wait_async0();
    __syncthreads();
  }

  const _Float16* Ap = A + (size_t)row0 * K;
  v8f acc[4] = {};
  const int kt = K >> 5;
  for (int kk = 0; kk < kt; ++kk) {
    v16h a = load_a16(Ap + kk * 32, K, lane);
    v16h bfr[4];
#pragma unroll
    for (int c = 0; c < 4; ++c)
      bfr[c] = load_bt16(smW + (size_t)(c * 16) * K + kk * 32, K, lane);
#pragma unroll
    for (int c = 0; c < 4; ++c)
      acc[c] = wmma_f16(a, bfr[c], acc[c]);
  }

  const int cn = lane & 15;
  const int mh = (lane >> 4) * 8;
#pragma unroll
  for (int c = 0; c < 4; ++c) {
    const int col = col0 + c * 16 + cn;
    const float bb = bias[col];
    if (mode == 2) {
      const int brow  = row0 / N_;
      const int nbase = (row0 % N_) + mh;
      _Float16 tmp[8];
#pragma unroll
      for (int g = 0; g < 8; ++g) tmp[g] = (_Float16)(acc[c][g] + bb);
      _Float16* vT = (_Float16*)outp;
      *(uint4*)(vT + ((size_t)brow * D_ + col) * N_ + nbase) = *(const uint4*)tmp;
    } else if (mode == 1) {
      float* of = (float*)outp;
#pragma unroll
      for (int g = 0; g < 8; ++g)
        of[(size_t)(row0 + mh + g) * 256 + col] = fmaxf(acc[c][g] + bb, 0.f);
    } else {
      _Float16* oh = (_Float16*)outp;
#pragma unroll
      for (int g = 0; g < 8; ++g)
        oh[(size_t)(row0 + mh + g) * 256 + col] = (_Float16)(acc[c][g] + bb);
    }
  }
}

// ---------------------------------------------------------------------------
// Attention: one block = 16 query rows for one (b, r).
// Phase 1: scores via WMMA -> LDS f32 (masked by adj), k prefetched ahead.
// Softmax per row (wave-shuffle reductions).
// Phase 2: out = alpha @ vT via WMMA (contiguous B loads from vT).
// grid(N/16, B, R), block(256), dynamic LDS = 16*1024*4 bytes.
// ---------------------------------------------------------------------------
__global__ void attn_kernel(const _Float16* __restrict__ q,
                            const _Float16* __restrict__ kmat,
                            const _Float16* __restrict__ vT,
                            const int*      __restrict__ adj,
                            _Float16*       __restrict__ attn) {
  extern __shared__ float sm[];                // 16 x N_ scores
  const int lane = threadIdx.x & 31;
  const int w    = threadIdx.x >> 5;
  const int row0 = blockIdx.x * 16;
  const int b    = blockIdx.y;
  const int r    = blockIdx.z;

  const _Float16* qp   = q    + ((size_t)b * N_ + row0) * D_;
  const _Float16* kp   = kmat + ((size_t)r * B_ + b) * (size_t)N_ * D_;
  const _Float16* vp   = vT   + ((size_t)r * B_ + b) * (size_t)D_ * N_;  // [d][n]
  const int*      adjp = adj  + (((size_t)r * B_ + b) * N_ + row0) * (size_t)N_;

  // q tile (16 x 256) as 8 A-fragments in registers
  v16h aq[8];
#pragma unroll
  for (int c = 0; c < 8; ++c) aq[c] = load_a16(qp + c * 32, D_, lane);

  const float scale = 0.0625f;                 // 1/sqrt(256)
  const int cn = lane & 15;
  const int mh = (lane >> 4) * 8;

  // ---- phase 1: scores. wave w owns j-tiles {w, w+8, ..., w+56} ----
  for (int t = 0; t < 8; ++t) {
    const int j = w + t * 8;
    if (t + 1 < 8)
      __builtin_prefetch(kp + (size_t)((w + (t + 1) * 8) * 16) * D_, 0, 1);
    const _Float16* kj = kp + (size_t)(j * 16) * D_;   // k rows are B-layout
    v8f acc = {};
#pragma unroll
    for (int c = 0; c < 8; ++c) {
      v16h bf = load_bt16(kj + c * 32, D_, lane);
      acc = wmma_f16(aq[c], bf, acc);
    }
#pragma unroll
    for (int g = 0; g < 8; ++g) {
      const int m = mh + g;
      const int n = j * 16 + cn;
      float s = acc[g] * scale;
      if (adjp[(size_t)m * N_ + n] == 0) s = -1e9f;
      sm[m * N_ + n] = s;
    }
  }
  __syncthreads();

  // ---- softmax: wave w owns rows {2w, 2w+1} ----
  for (int rr = 0; rr < 2; ++rr) {
    float* rowp = sm + (w * 2 + rr) * N_;
    float mx = -1e30f;
    for (int i = lane; i < N_; i += 32) mx = fmaxf(mx, rowp[i]);
#pragma unroll
    for (int off = 16; off >= 1; off >>= 1) mx = fmaxf(mx, __shfl_xor(mx, off, 32));
    float ssum = 0.f;
    for (int i = lane; i < N_; i += 32) {
      const float e = __expf(rowp[i] - mx);
      rowp[i] = e;
      ssum += e;
    }
#pragma unroll
    for (int off = 16; off >= 1; off >>= 1) ssum += __shfl_xor(ssum, off, 32);
    const float inv = 1.0f / ssum;
    for (int i = lane; i < N_; i += 32) rowp[i] *= inv;
  }
  __syncthreads();

  // ---- phase 2: out = alpha @ v. wave w owns col tiles {2w, 2w+1};
  //      one LDS A-fragment feeds both tiles. ----
  {
    const int ct0 = w * 2, ct1 = w * 2 + 1;
    v8f acc0 = {}, acc1 = {};
    for (int it = 0; it < N_ / 32; ++it) {
      v16h a  = load_a_lds32(sm + it * 32, N_, lane);
      v16h b0 = load_bt16(vp + (size_t)(ct0 * 16) * N_ + it * 32, N_, lane);
      v16h b1 = load_bt16(vp + (size_t)(ct1 * 16) * N_ + it * 32, N_, lane);
      acc0 = wmma_f16(a, b0, acc0);
      acc1 = wmma_f16(a, b1, acc1);
    }
#pragma unroll
    for (int g = 0; g < 8; ++g) {
      const int m = mh + g;
      const size_t base = ((size_t)b * N_ + row0 + m) * RD_ + (size_t)r * D_;
      attn[base + ct0 * 16 + cn] = (_Float16)acc0[g];
      attn[base + ct1 * 16 + cn] = (_Float16)acc1[g];
    }
  }
}

// ---------------------------------------------------------------------------
// Residual + LayerNorm: out = LN(h + hfused) * g + b.   grid(B*N), block(D)
// ---------------------------------------------------------------------------
__global__ void ln_kernel(const float* __restrict__ h, const float* __restrict__ hf,
                          const float* __restrict__ g,  const float* __restrict__ bta,
                          float* __restrict__ out) {
  __shared__ float red[D_];
  const size_t row = blockIdx.x;
  const int d = threadIdx.x;
  const float v = h[row * D_ + d] + hf[row * D_ + d];
  red[d] = v;
  __syncthreads();
  for (int s = D_ / 2; s > 0; s >>= 1) { if (d < s) red[d] += red[d + s]; __syncthreads(); }
  const float mu = red[0] * (1.0f / D_);
  __syncthreads();
  const float c = v - mu;
  red[d] = c * c;
  __syncthreads();
  for (int s = D_ / 2; s > 0; s >>= 1) { if (d < s) red[d] += red[d + s]; __syncthreads(); }
  const float var = red[0] * (1.0f / D_);
  out[row * D_ + d] = c * rsqrtf(var + 1e-5f) * g[d] + bta[d];
}

// ---------------------------------------------------------------------------
// Host launcher
// ---------------------------------------------------------------------------
extern "C" void kernel_launch(void* const* d_in, const int* in_sizes, int n_in,
                              void* d_out, int out_size, void* d_ws, size_t ws_size,
                              hipStream_t stream) {
  const float* node_embed = (const float*)d_in[0];
  const float* input_x    = (const float*)d_in[1];
  const int*   adj        = (const int*)  d_in[2];
  const float* WQi_w      = (const float*)d_in[3];
  const float* WQi_b      = (const float*)d_in[4];
  const float* WQ_w       = (const float*)d_in[5];
  const float* WQ_b       = (const float*)d_in[6];
  const float* WK_w       = (const float*)d_in[7];
  const float* WK_b       = (const float*)d_in[8];
  const float* WV_w       = (const float*)d_in[9];
  const float* WV_b       = (const float*)d_in[10];
  const float* WF_w       = (const float*)d_in[11];
  const float* WF_b       = (const float*)d_in[12];
  const float* ln_g       = (const float*)d_in[13];
  const float* ln_b       = (const float*)d_in[14];
  float* out = (float*)d_out;

  const size_t BN = (size_t)B_ * N_;

  char* wp = (char*)d_ws;
  auto alloc = [&](size_t bytes) -> char* {
    char* p = wp;
    wp += (bytes + 255) & ~(size_t)255;
    return p;
  };
  _Float16* qc     = (_Float16*)alloc(BN * DF_ * sizeof(_Float16));
  _Float16* q16    = (_Float16*)alloc(BN * D_ * sizeof(_Float16));
  _Float16* k16    = (_Float16*)alloc((size_t)R_ * BN * D_ * sizeof(_Float16));
  _Float16* vT16   = (_Float16*)alloc((size_t)R_ * BN * D_ * sizeof(_Float16)); // [r][b][d][n]
  _Float16* attn16 = (_Float16*)alloc(BN * RD_ * sizeof(_Float16));
  float*    hfused = (float*)   alloc(BN * D_ * sizeof(float));
  float*    h_ws   = (float*)   alloc(BN * D_ * sizeof(float));
  float*    x1     = (float*)   alloc((size_t)B_ * F_ * sizeof(float));
  _Float16* wq_t   = (_Float16*)alloc((size_t)D_ * DF_ * sizeof(_Float16));
  _Float16* wk_t   = (_Float16*)alloc((size_t)R_ * D_ * DF_ * sizeof(_Float16));
  _Float16* wv_t   = (_Float16*)alloc((size_t)R_ * D_ * DF_ * sizeof(_Float16));
  _Float16* wf_t   = (_Float16*)alloc((size_t)D_ * RD_ * sizeof(_Float16));

  const dim3 gemm_grid(128, 4);                 // 16384/128 rows, 256/64 cols
  const size_t lds_df = (size_t)64 * DF_ * sizeof(_Float16);  // 48 KB
  const size_t lds_rd = (size_t)64 * RD_ * sizeof(_Float16);  // 64 KB
  const int tc_blk = 256;

  for (int l = 0; l < L_; ++l) {
    const float* hsrc = (l == 0) ? node_embed : h_ws;
    float* hdst = (l == L_ - 1) ? out : h_ws;

    x1_kernel<<<B_, F_, 0, stream>>>(input_x, WQi_w + (size_t)l * F_ * F_,
                                     WQi_b + (size_t)l * F_, x1);
    build_qc_kernel<<<(unsigned)BN, DF_, 0, stream>>>(hsrc, x1, qc);

    {
      const size_t ndf = (size_t)DF_ * 256, nrd = (size_t)RD_ * 256;
      tconv_kernel<<<(unsigned)((ndf + tc_blk - 1) / tc_blk), tc_blk, 0, stream>>>(
          WQ_w + (size_t)l * DF_ * D_, wq_t, DF_);
      for (int r = 0; r < R_; ++r) {
        tconv_kernel<<<(unsigned)((ndf + tc_blk - 1) / tc_blk), tc_blk, 0, stream>>>(
            WK_w + ((size_t)l * R_ + r) * DF_ * D_, wk_t + (size_t)r * D_ * DF_, DF_);
        tconv_kernel<<<(unsigned)((ndf + tc_blk - 1) / tc_blk), tc_blk, 0, stream>>>(
            WV_w + ((size_t)l * R_ + r) * DF_ * D_, wv_t + (size_t)r * D_ * DF_, DF_);
      }
      tconv_kernel<<<(unsigned)((nrd + tc_blk - 1) / tc_blk), tc_blk, 0, stream>>>(
          WF_w + (size_t)l * RD_ * D_, wf_t, RD_);
    }

    // projections: q (mode 0), k[r] (mode 0), v[r] (mode 2 -> transposed)
    wmma_gemm_kernel<<<gemm_grid, 256, lds_df, stream>>>(
        qc, wq_t, WQ_b + (size_t)l * D_, q16, DF_, 0);
    for (int r = 0; r < R_; ++r) {
      wmma_gemm_kernel<<<gemm_grid, 256, lds_df, stream>>>(
          qc, wk_t + (size_t)r * D_ * DF_, WK_b + ((size_t)l * R_ + r) * D_,
          k16 + (size_t)r * BN * D_, DF_, 0);
      wmma_gemm_kernel<<<gemm_grid, 256, lds_df, stream>>>(
          qc, wv_t + (size_t)r * D_ * DF_, WV_b + ((size_t)l * R_ + r) * D_,
          vT16 + (size_t)r * BN * D_, DF_, 2);
    }

    attn_kernel<<<dim3(N_ / 16, B_, R_), 256, 16 * N_ * sizeof(float), stream>>>(
        q16, k16, vT16, adj, attn16);

    // fuse GEMM with bias + ReLU -> f32 (mode 1)
    wmma_gemm_kernel<<<gemm_grid, 256, lds_rd, stream>>>(
        attn16, wf_t, WF_b + (size_t)l * D_, hfused, RD_, 1);

    ln_kernel<<<(unsigned)BN, D_, 0, stream>>>(hsrc, hfused,
                                               ln_g + (size_t)l * D_,
                                               ln_b + (size_t)l * D_, hdst);
  }
}
